// VQVAE_18159121727588
// MI455X (gfx1250) — matrix operand order repaired and checked by reading, not compiled, not run on hardware
//
#include <hip/hip_runtime.h>
#include <hip/hip_bf16.h>

typedef __attribute__((ext_vector_type(16))) __bf16 v16bf;
typedef __attribute__((ext_vector_type(8)))  float  v8f;

#define DEV __device__ __forceinline__

// WMMA 16x16x32 bf16: A-matrix per-lane element i -> semantic k.
DEV int kofA(int i, int half) {
  return ((i >> 3) * 16) + (((i >> 1) & 3) * 2) + (i & 1) + half * 8;
}
// B-matrix: same k order as the A-matrix.
DEV int kofB(int i, int half) { return kofA(i, half); }

// ---------------------------------------------------------------------------
// Prep: permute an [M][K] f32 matrix into WMMA-native bf16 A-fragment layout.
// out[(((mt*(K/32)+kt)*32 + lane)*16 + i] = in[(mt*16+(lane&15))*K + kt*32 + kofA(i,lane>>4)]
// -> each lane's A fragment is one contiguous 32B vector load.
// ---------------------------------------------------------------------------
__global__ void permA_k(const float* __restrict__ in, __bf16* __restrict__ out,
                        int M, int K) {
  int idx = blockIdx.x * blockDim.x + threadIdx.x;   // M*K elements
  if (idx >= M * K) return;
  int i = idx & 15;
  int lane = (idx >> 4) & 31;
  int t = idx >> 9;                 // mt*(K/32) + kt
  int KT = K >> 5;
  int kt = t % KT, mt = t / KT;
  int row = mt * 16 + (lane & 15);
  int col = kt * 32 + kofA(i, lane >> 4);
  out[idx] = (__bf16)in[row * K + col];
}

// ConvTranspose weights -> per-parity-class, WMMA-permuted bf16 fragments.
__global__ void prep_wtT_k(const float* __restrict__ w, __bf16* __restrict__ out) {
  int idx = blockIdx.x * blockDim.x + threadIdx.x;   // 8 * 32*512 = 131072
  if (idx >= 131072) return;
  int cls = idx >> 14, rem = idx & 16383;
  int i = rem & 15, lane = (rem >> 4) & 31, t = rem >> 9;  // t = mt*16 + kt (KT=16)
  int kt = t & 15, mt = t >> 4;
  int pd = (cls >> 2) & 1, ph = (cls >> 1) & 1, pw = cls & 1;
  int co = mt * 16 + (lane & 15);
  int k = kt * 32 + kofA(i, lane >> 4);
  int ci = k >> 3, tt = k & 7;
  int td = (1 - pd) + 2 * ((tt >> 2) & 1);
  int th = (1 - ph) + 2 * ((tt >> 1) & 1);
  int tw = (1 - pw) + 2 * (tt & 1);
  // dec_wt shape (Cin=64, Cout=32, 4,4,4)
  out[idx] = (__bf16)w[((ci * 32 + co) << 6) + (td << 4) + (th << 2) + tw];
}

__global__ void codebook_norms_k(const float* __restrict__ cb, float* __restrict__ hn) {
  int k = blockIdx.x * blockDim.x + threadIdx.x;
  if (k >= 512) return;
  float s = 0.0f;
  for (int c = 0; c < 64; ++c) { float v = cb[k * 64 + c]; s += v * v; }
  hn[k] = 0.5f * s;
}

__global__ void zero_red_k(float* loss, int* cnt) {
  if (threadIdx.x == 0) loss[0] = 0.0f;
  if (threadIdx.x < 512) cnt[threadIdx.x] = 0;
}

// ---------------------------------------------------------------------------
// Encoder conv1: Cin=1, k=4, s=2, p=1, fused BN+ReLU (VALU; <1% of FLOPs)
// ---------------------------------------------------------------------------
__global__ __launch_bounds__(256)
void enc_conv1_k(const float* __restrict__ x, const float* __restrict__ w,
                 const float* __restrict__ bias, const float* __restrict__ g,
                 const float* __restrict__ be, const float* __restrict__ mu,
                 const float* __restrict__ var, __bf16* __restrict__ h1) {
  int i = blockIdx.x * 256 + threadIdx.x;        // 8*32*32768
  int sp = i & 32767; int co = (i >> 15) & 31; int b = i >> 20;
  int z0 = sp >> 10, y0 = (sp >> 5) & 31, x0 = sp & 31;
  float acc = 0.0f;
#pragma unroll
  for (int td = 0; td < 4; ++td) {
    int zz = z0 * 2 - 1 + td;
#pragma unroll
    for (int th = 0; th < 4; ++th) {
      int yy = y0 * 2 - 1 + th;
#pragma unroll
      for (int tw = 0; tw < 4; ++tw) {
        int xx = x0 * 2 - 1 + tw;
        bool ok = ((unsigned)zz < 64u) & ((unsigned)yy < 64u) & ((unsigned)xx < 64u);
        float v = x[(b << 18) + ((zz & 63) << 12) + ((yy & 63) << 6) + (xx & 63)];
        acc = fmaf(ok ? v : 0.0f, w[(co << 6) + (td << 4) + (th << 2) + tw], acc);
      }
    }
  }
  float s = g[co] * rsqrtf(var[co] + 1e-5f);
  float v = (acc + bias[co]) * s + (be[co] - mu[co] * s);
  h1[i] = (__bf16)fmaxf(v, 0.0f);
}

// ---------------------------------------------------------------------------
// Implicit-GEMM 3x3x3 conv (stride1, pad1) over 32^3 x B=8 via WMMA bf16.
// One wave -> 16(Cout) x 16(positions) tile; A from pre-permuted fragments
// (one v16bf load), B gathered unconditionally with clamp+select (no branches).
// ---------------------------------------------------------------------------
template <int CIN, int COUT, bool BN, bool RELU, bool F32OUT>
__global__ __launch_bounds__(256)
void conv3x3_wmma_k(const __bf16* __restrict__ in, const __bf16* __restrict__ wp,
                    const float* __restrict__ bias,
                    const float* __restrict__ g, const float* __restrict__ be,
                    const float* __restrict__ mu, const float* __restrict__ var,
                    __bf16* __restrict__ outb, float* __restrict__ outf) {
  constexpr int K = CIN * 27;
  constexpr int KT = K / 32;
  constexpr int COT = COUT / 16;
  const int lane = threadIdx.x & 31;
  const int wave = threadIdx.x >> 5;
  const int tile = blockIdx.x * 8 + wave;
  const int posTile = tile / COT;
  const int coTile  = tile % COT;
  const int half = (lane >> 4) & 1;
  const int mr = lane & 15;

  const int nn = posTile * 16 + mr;   // this lane's output column
  const int b  = nn >> 15;
  const int sp = nn & 32767;
  const int z0 = sp >> 10, y0 = (sp >> 5) & 31, x0 = sp & 31;
  const __bf16 zero = (__bf16)0.0f;

  // lane's A-fragment stream: contiguous 32B per k-step
  const __bf16* afrag = wp + (((size_t)(coTile * KT) * 32 + lane) << 4);

  v8f acc = {};
  for (int kt = 0; kt < KT; ++kt) {
    __builtin_prefetch(afrag + ((kt + 1) << 9), 0, 1);
    v16bf a = *(const v16bf*)(afrag + ((size_t)kt << 9));   // step = 32 lanes *16
    v16bf bm;
    const int k0 = kt * 32 + half * 8;
#pragma unroll
    for (int i = 0; i < 16; ++i) {
      int kb = k0 + (i & 7) + ((i >> 3) << 4);
      int ci = kb / 27, t = kb - ci * 27;
      int td = t / 9, rr = t - td * 9;
      int th = rr / 3, tw = rr - th * 3;
      int zz = z0 + td - 1, yy = y0 + th - 1, xx = x0 + tw - 1;
      bool ok = ((unsigned)zz < 32u) & ((unsigned)yy < 32u) & ((unsigned)xx < 32u);
      __bf16 v = in[((b * CIN + ci) << 15) + ((zz & 31) << 10) + ((yy & 31) << 5) + (xx & 31)];
      bm[i] = ok ? v : zero;
    }
    acc = __builtin_amdgcn_wmma_f32_16x16x32_bf16(false, a, false, bm,
                                                  (short)0, acc, false, false);
  }
#pragma unroll
  for (int r = 0; r < 8; ++r) {
    int co = coTile * 16 + r + half * 8;
    float v = acc[r] + bias[co];
    if (BN) {
      float s = g[co] * rsqrtf(var[co] + 1e-5f);
      v = v * s + (be[co] - mu[co] * s);
    }
    if (RELU) v = fmaxf(v, 0.0f);
    int off = ((b * COUT + co) << 15) + sp;
    if (F32OUT) outf[off] = v; else outb[off] = (__bf16)v;
  }
}

// ---------------------------------------------------------------------------
// ConvTranspose3d k=4 s=2 p=1 as 8 parity-class implicit GEMMs (K=64*8=512),
// fused BN+ReLU, WMMA bf16. Output (8,32,64^3) bf16.
// ---------------------------------------------------------------------------
__global__ __launch_bounds__(256)
void convT_wmma_k(const __bf16* __restrict__ d1, const __bf16* __restrict__ wt,
                  const float* __restrict__ bt, const float* __restrict__ g,
                  const float* __restrict__ be, const float* __restrict__ mu,
                  const float* __restrict__ var, __bf16* __restrict__ ht) {
  const int lane = threadIdx.x & 31;
  const int wave = threadIdx.x >> 5;
  int tile = blockIdx.x * 8 + wave;          // cls*16384*2 + posTile*2 + coTile
  const int coTile = tile & 1; tile >>= 1;
  const int posTile = tile & 16383;
  const int cls = tile >> 14;                // 0..7
  const int pd = (cls >> 2) & 1, ph = (cls >> 1) & 1, pw = cls & 1;
  const int half = (lane >> 4) & 1;
  const int mr = lane & 15;
  const int nn = posTile * 16 + mr;
  const int b = nn >> 15, sp = nn & 32767;
  const int qd = sp >> 10, qh = (sp >> 5) & 31, qw = sp & 31;
  const __bf16 zero = (__bf16)0.0f;

  // permuted per-class fragments: class block = 32*512 elems; KT=16, mt=coTile
  const __bf16* afrag = wt + ((size_t)cls << 14) + (((size_t)(coTile * 16) * 32 + lane) << 4);

  v8f acc = {};
  for (int kt = 0; kt < 16; ++kt) {
    __builtin_prefetch(afrag + ((kt + 1) << 9), 0, 1);
    v16bf a = *(const v16bf*)(afrag + ((size_t)kt << 9));
    v16bf bm;
    const int k0 = kt * 32 + half * 8;
#pragma unroll
    for (int i = 0; i < 16; ++i) {
      int kb = k0 + (i & 7) + ((i >> 3) << 4);
      int ci = kb >> 3, tt = kb & 7;
      int id = qd + pd - ((tt >> 2) & 1);
      int ih = qh + ph - ((tt >> 1) & 1);
      int iw = qw + pw - (tt & 1);
      bool ok = ((unsigned)id < 32u) & ((unsigned)ih < 32u) & ((unsigned)iw < 32u);
      __bf16 v = d1[((b * 64 + ci) << 15) + ((id & 31) << 10) + ((ih & 31) << 5) + (iw & 31)];
      bm[i] = ok ? v : zero;
    }
    acc = __builtin_amdgcn_wmma_f32_16x16x32_bf16(false, a, false, bm,
                                                  (short)0, acc, false, false);
  }
  const int od = (qd << 1) + pd, oh = (qh << 1) + ph, ow = (qw << 1) + pw;
#pragma unroll
  for (int r = 0; r < 8; ++r) {
    int co = coTile * 16 + r + half * 8;
    float s = g[co] * rsqrtf(var[co] + 1e-5f);
    float v = acc[r] + bt[co];
    v = fmaxf(v * s + (be[co] - mu[co] * s), 0.0f);
    ht[((size_t)(b * 32 + co) << 18) + (od << 12) + (oh << 6) + ow] = (__bf16)v;
  }
}

// ---------------------------------------------------------------------------
// VQ argmax: scores = z.c - 0.5|c|^2 via WMMA; pre-permuted codebook staged
// in 64KB LDS so A-fragments are two ds_load_b128 per step.
// ---------------------------------------------------------------------------
__global__ __launch_bounds__(256)
void vq_argmax_k(const float* __restrict__ zf, const __bf16* __restrict__ cbp,
                 const float* __restrict__ hn, int* __restrict__ idx) {
  __shared__ __attribute__((aligned(32))) __bf16 lcb[512 * 64];   // 65536 B
  for (int i = threadIdx.x; i < 512 * 64; i += 256) lcb[i] = cbp[i];
  __syncthreads();
  const int lane = threadIdx.x & 31, wave = threadIdx.x >> 5;
  const int half = (lane >> 4) & 1, mr = lane & 15;
  const int posTile = blockIdx.x * 8 + wave;
  const int nn = posTile * 16 + mr;
  const int b = nn >> 15, sp = nn & 32767;
  v16bf b0, b1;
#pragma unroll
  for (int i = 0; i < 16; ++i) {
    int k = kofB(i, half);
    b0[i] = (__bf16)zf[((b * 64 + k) << 15) + sp];
    b1[i] = (__bf16)zf[((b * 64 + k + 32) << 15) + sp];
  }
  float best = -3.4e38f; int bi = 0;
  for (int ct = 0; ct < 32; ++ct) {
    // permuted layout: frag off = ((ct*2 + kt)*32 + lane)*16
    v16bf a0 = *(const v16bf*)&lcb[(((ct * 2 + 0) * 32 + lane) << 4)];
    v16bf a1 = *(const v16bf*)&lcb[(((ct * 2 + 1) * 32 + lane) << 4)];
    v8f c = {};
    c = __builtin_amdgcn_wmma_f32_16x16x32_bf16(false, a0, false, b0, (short)0, c, false, false);
    c = __builtin_amdgcn_wmma_f32_16x16x32_bf16(false, a1, false, b1, (short)0, c, false, false);
#pragma unroll
    for (int r = 0; r < 8; ++r) {
      int code = ct * 16 + r + half * 8;
      float s = c[r] - hn[code];
      if (s > best) { best = s; bi = code; }
    }
  }
  // combine the two wave halves (codes split 8-apart across halves)
  float ob = __shfl_xor(best, 16);
  int   oi = __shfl_xor(bi, 16);
  if (ob > best || (ob == best && oi < bi)) { best = ob; bi = oi; }
  if (lane < 16) idx[nn] = bi;
}

// ---------------------------------------------------------------------------
// Quantize + commit loss partials + code histogram
// ---------------------------------------------------------------------------
__global__ __launch_bounds__(256)
void vq_quant_k(const float* __restrict__ zf, const float* __restrict__ cb,
                const int* __restrict__ idx, __bf16* __restrict__ q,
                float* __restrict__ loss, int* __restrict__ cnt) {
  __shared__ float red[256];
  int n = blockIdx.x * 256 + threadIdx.x;     // 262144 positions
  int b = n >> 15, sp = n & 32767;
  int id = idx[n];
  atomicAdd(&cnt[id], 1);
  float s = 0.0f;
  for (int c = 0; c < 64; ++c) {
    float zv = zf[((b * 64 + c) << 15) + sp];
    float cv = cb[(id << 6) + c];
    float d = zv - cv; s += d * d;
    q[((b * 64 + c) << 15) + sp] = (__bf16)cv;
  }
  red[threadIdx.x] = s;
  __syncthreads();
  for (int off = 128; off > 0; off >>= 1) {
    if (threadIdx.x < off) red[threadIdx.x] += red[threadIdx.x + off];
    __syncthreads();
  }
  if (threadIdx.x == 0) atomicAdd(loss, red[0]);
}

__global__ void vq_final_k(const float* __restrict__ loss, const int* __restrict__ cnt,
                           float* __restrict__ out) {
  __shared__ float red[512];
  int k = threadIdx.x;
  float p = cnt[k] * (1.0f / 262144.0f);
  red[k] = p * logf(p + 1e-10f);
  __syncthreads();
  for (int off = 256; off > 0; off >>= 1) {
    if (k < off) red[k] += red[k + off];
    __syncthreads();
  }
  if (k == 0) {
    out[2097152] = 0.25f * loss[0] / 16777216.0f;  // vq_loss
    out[2097153] = expf(-red[0]);                   // perplexity
  }
}

// ---------------------------------------------------------------------------
// Decoder conv3: Cin=32 -> Cout=1, k=3, pad=1 over 64^3, sigmoid epilogue
// ---------------------------------------------------------------------------
__global__ __launch_bounds__(256)
void dec_conv3_k(const __bf16* __restrict__ ht, const float* __restrict__ w,
                 const float* __restrict__ bias, float* __restrict__ out) {
  int i = blockIdx.x * 256 + threadIdx.x;      // 8 * 262144
  int sp = i & 262143; int b = i >> 18;
  int z0 = sp >> 12, y0 = (sp >> 6) & 63, x0 = sp & 63;
  float acc = 0.0f;
  for (int ci = 0; ci < 32; ++ci) {
    const __bf16* base = ht + ((size_t)(b * 32 + ci) << 18);
    const float* wc = w + ci * 27;
#pragma unroll
    for (int t = 0; t < 27; ++t) {
      int td = t / 9, rr = t - td * 9, th = rr / 3, tw = rr - th * 3;
      int zz = z0 + td - 1, yy = y0 + th - 1, xx = x0 + tw - 1;
      bool ok = ((unsigned)zz < 64u) & ((unsigned)yy < 64u) & ((unsigned)xx < 64u);
      float v = (float)base[((zz & 63) << 12) + ((yy & 63) << 6) + (xx & 63)];
      acc = fmaf(ok ? v : 0.0f, wc[t], acc);
    }
  }
  float v = acc + bias[0];
  out[i] = 1.0f / (1.0f + expf(-v));
}

// ---------------------------------------------------------------------------
// Host launcher
// ---------------------------------------------------------------------------
extern "C" void kernel_launch(void* const* d_in, const int* in_sizes, int n_in,
                              void* d_out, int out_size, void* d_ws, size_t ws_size,
                              hipStream_t stream) {
  (void)in_sizes; (void)n_in; (void)out_size; (void)ws_size;
  const float* x    = (const float*)d_in[0];
  const float* cb   = (const float*)d_in[1];
  const float* ew1  = (const float*)d_in[2];  const float* eb1  = (const float*)d_in[3];
  const float* eg1  = (const float*)d_in[4];  const float* ebe1 = (const float*)d_in[5];
  const float* em1  = (const float*)d_in[6];  const float* ev1  = (const float*)d_in[7];
  const float* ew2  = (const float*)d_in[8];  const float* eb2  = (const float*)d_in[9];
  const float* eg2  = (const float*)d_in[10]; const float* ebe2 = (const float*)d_in[11];
  const float* em2  = (const float*)d_in[12]; const float* ev2  = (const float*)d_in[13];
  const float* ew3  = (const float*)d_in[14]; const float* eb3  = (const float*)d_in[15];
  const float* dw1  = (const float*)d_in[16]; const float* db1  = (const float*)d_in[17];
  const float* dg1  = (const float*)d_in[18]; const float* dbe1 = (const float*)d_in[19];
  const float* dm1  = (const float*)d_in[20]; const float* dv1  = (const float*)d_in[21];
  const float* dwt  = (const float*)d_in[22]; const float* dbt  = (const float*)d_in[23];
  const float* dg2  = (const float*)d_in[24]; const float* dbe2 = (const float*)d_in[25];
  const float* dm2  = (const float*)d_in[26]; const float* dv2  = (const float*)d_in[27];
  const float* dw3  = (const float*)d_in[28]; const float* db3  = (const float*)d_in[29];

  char* ws = (char*)d_ws;
  // Workspace layout (overlaid: ht reuses h1/h2/z/quant region after they die)
  constexpr size_t OFF_H1 = 0;                       // bf16 8*32*32^3  (16 MiB)
  constexpr size_t OFF_H2 = 16777216;                // bf16 8*64*32^3  (32 MiB)
  constexpr size_t OFF_Z  = 50331648;                // f32  8*64*32^3  (64 MiB)
  constexpr size_t OFF_Q  = 117440512;               // bf16 8*64*32^3  (32 MiB)
  constexpr size_t OFF_D1 = 150994944;               // bf16 8*64*32^3  (32 MiB)
  constexpr size_t OFF_HT = 0;                       // bf16 8*32*64^3 (128 MiB, overlays dead bufs)
  constexpr size_t OFF_WB = 184549376;               // weights etc.
  __bf16* h1   = (__bf16*)(ws + OFF_H1);
  __bf16* h2   = (__bf16*)(ws + OFF_H2);
  float*  zf   = (float*)(ws + OFF_Z);
  __bf16* qnt  = (__bf16*)(ws + OFF_Q);
  __bf16* d1b  = (__bf16*)(ws + OFF_D1);
  __bf16* htb  = (__bf16*)(ws + OFF_HT);
  __bf16* wb2  = (__bf16*)(ws + OFF_WB);             // 64*864  permuted
  __bf16* wb3  = (__bf16*)(ws + OFF_WB + 131072);    // 64*1728 permuted
  __bf16* wbd1 = (__bf16*)(ws + OFF_WB + 393216);    // 64*1728 permuted
  __bf16* wtb  = (__bf16*)(ws + OFF_WB + 655360);    // 8*32*512 permuted
  __bf16* cbp  = (__bf16*)(ws + OFF_WB + 917504);    // 512*64  permuted
  float*  hn   = (float*)(ws + OFF_WB + 983040);     // 512
  int*    idx  = (int*)(ws + OFF_WB + 1048576);      // 262144
  float*  loss = (float*)(ws + OFF_WB + 2097152);
  int*    cnt  = (int*)(ws + OFF_WB + 2097408);      // 512
  float*  out  = (float*)d_out;

  // ---- prep: WMMA-permuted bf16 weights, codebook norms, zero reductions ----
  permA_k<<<(55296 + 255) / 256, 256, 0, stream>>>(ew2, wb2, 64, 864);
  permA_k<<<(110592 + 255) / 256, 256, 0, stream>>>(ew3, wb3, 64, 1728);
  permA_k<<<(110592 + 255) / 256, 256, 0, stream>>>(dw1, wbd1, 64, 1728);
  prep_wtT_k<<<131072 / 256, 256, 0, stream>>>(dwt, wtb);
  permA_k<<<(32768 + 255) / 256, 256, 0, stream>>>(cb, cbp, 512, 64);
  codebook_norms_k<<<2, 256, 0, stream>>>(cb, hn);
  zero_red_k<<<1, 512, 0, stream>>>(loss, cnt);

  // ---- encoder ----
  enc_conv1_k<<<32768, 256, 0, stream>>>(x, ew1, eb1, eg1, ebe1, em1, ev1, h1);
  conv3x3_wmma_k<32, 64, true, true, false><<<8192, 256, 0, stream>>>(
      h1, wb2, eb2, eg2, ebe2, em2, ev2, h2, nullptr);
  conv3x3_wmma_k<64, 64, false, false, true><<<8192, 256, 0, stream>>>(
      h2, wb3, eb3, nullptr, nullptr, nullptr, nullptr, nullptr, zf);

  // ---- vector quantizer ----
  vq_argmax_k<<<2048, 256, 0, stream>>>(zf, cbp, hn, idx);
  vq_quant_k<<<1024, 256, 0, stream>>>(zf, cb, idx, qnt, loss, cnt);
  vq_final_k<<<1, 512, 0, stream>>>(loss, cnt, out);

  // ---- decoder ----
  conv3x3_wmma_k<64, 64, true, true, false><<<8192, 256, 0, stream>>>(
      qnt, wbd1, db1, dg1, dbe1, dm1, dv1, d1b, nullptr);
  convT_wmma_k<<<32768, 256, 0, stream>>>(d1b, wtb, dbt, dg2, dbe2, dm2, dv2, htb);
  dec_conv3_k<<<8192, 256, 0, stream>>>(htb, dw3, db3, out);
}
